// SRNN_45758581571881
// MI455X (gfx1250) — matrix operand order repaired and checked by reading, not compiled, not run on hardware
//
#include <hip/hip_runtime.h>

// SRNN forward on MI455X (gfx1250), wave32 + bf16 WMMA (f32 accumulate).
// Per-timestep kernel pair (recurrent+input WMMA GEMM, then f32 output
// projection), all launched on `stream` so the harness can graph-capture.

#define N_IN   700
#define N_INP  704          // padded to multiple of 32 (WMMA K step)
#define N_REC  1024
#define N_OUT  20
#define N_T    500
#define N_B    128
#define THRv   1.0f
#define ALPHAv 0.951229424500714f   // exp(-1/20)
#define KAPPAv 0.951229424500714f

typedef __attribute__((ext_vector_type(16))) __bf16 v16bf;
typedef __attribute__((ext_vector_type(8)))  __bf16 v8bf;
typedef __attribute__((ext_vector_type(8)))  float  v8f;

union AFrag { v16bf v; struct { v8bf lo, hi; } h; };

// A-matrix fragment (16xK tile, contiguous k order matching B):
// lanes 0-15: halves 0..15 = A[m][k0..k0+15]
// lanes16-31: halves 0..15 = A[m][k0+16..k0+31]
__device__ __forceinline__ v16bf load_a_frag(const __bf16* __restrict__ base,
                                             int lda, int m0, int k0, int lane) {
  const int m  = m0 + (lane & 15);
  const int kb = k0 + ((lane >> 4) << 4);       // +16 for hi half-wave
  const __bf16* p = base + m * lda + kb;
  AFrag f;
  f.h.lo = *reinterpret_cast<const v8bf*>(p);
  f.h.hi = *reinterpret_cast<const v8bf*>(p + 8);
  return f.v;
}

// B-matrix fragment (Kx16). B = W^T so B[k][n] = W[j0+n][k]: one contiguous
// 16-element (32B) load per lane. lanes 0-15: K=k0..k0+15; lanes16-31: +16.
__device__ __forceinline__ v16bf load_b_frag(const __bf16* __restrict__ w,
                                             int ldw, int j0, int k0, int lane) {
  const int n  = j0 + (lane & 15);
  const int kb = k0 + ((lane >> 4) << 4);       // +16 for hi half-wave
  return *reinterpret_cast<const v16bf*>(w + n * ldw + kb);
}

// One LIF timestep: v_new = a*v + z@Wrec^T + x_t@Win^T - z*THR ; z_new = v_new>THR
// Grid: 64 blocks (N tiles) x 256 threads (8 waves = M tiles). One 16x16 C
// tile per wave. v updated in place (tile-owned); z double-buffered.
__global__ __launch_bounds__(256)
void srnn_step(const __bf16* __restrict__ xt,     // [128][704] bf16 (padded)
               const __bf16* __restrict__ winb,   // [1024][704] bf16
               const __bf16* __restrict__ wrecb,  // [1024][1024] bf16, diag=0
               const __bf16* __restrict__ zcur,   // [128][1024] bf16 spikes
               float*        __restrict__ v,      // [128][1024] f32 membrane
               __bf16*       __restrict__ znext)  // [128][1024] bf16 spikes out
{
  const int lane = threadIdx.x & 31;
  const int wave = threadIdx.x >> 5;
  const int m0 = wave * 16;
  const int j0 = blockIdx.x * 16;

  v8f acc = {};
  // input projection: K = 704
  #pragma unroll 2
  for (int k0 = 0; k0 < N_INP; k0 += 32) {
    v16bf a = load_a_frag(xt, N_INP, m0, k0, lane);
    v16bf b = load_b_frag(winb, N_INP, j0, k0, lane);
    acc = __builtin_amdgcn_wmma_f32_16x16x32_bf16(false, a, false, b,
                                                  (short)0, acc, false, false);
  }
  // recurrent projection: K = 1024
  #pragma unroll 2
  for (int k0 = 0; k0 < N_REC; k0 += 32) {
    v16bf a = load_a_frag(zcur, N_REC, m0, k0, lane);
    v16bf b = load_b_frag(wrecb, N_REC, j0, k0, lane);
    acc = __builtin_amdgcn_wmma_f32_16x16x32_bf16(false, a, false, b,
                                                  (short)0, acc, false, false);
  }

  // C layout: VGPR r -> row m0 + r (+8 for lanes 16-31), col j0 + (lane&15)
  const int col = j0 + (lane & 15);
  const int mb  = m0 + ((lane >> 4) << 3);
  #pragma unroll
  for (int r = 0; r < 8; ++r) {
    const int idx = (mb + r) * N_REC + col;
    const float zold = (float)zcur[idx];
    const float vnew = ALPHAv * v[idx] + acc[r] - THRv * zold;
    v[idx] = vnew;
    znext[idx] = (__bf16)((vnew > THRv) ? 1.0f : 0.0f);
  }
}

// Output projection (full f32): vo = KAPPA*vo + z_new @ w_out^T, write slice.
// Grid: 128 blocks (batch) x 640 threads (20 waves -> one output each).
__global__ __launch_bounds__(640)
void srnn_out(const __bf16* __restrict__ z,     // [128][1024]
              const float*  __restrict__ wout,  // [20][1024]
              float*        __restrict__ vo,    // [128][20] state
              float*        __restrict__ outp)  // [128][20] slice of d_out
{
  const int b = blockIdx.x;
  const int o = threadIdx.x >> 5;
  const int lane = threadIdx.x & 31;
  const __bf16* zr = z + b * N_REC;
  const float*  wr = wout + o * N_REC;
  float s = 0.f;
  #pragma unroll 4
  for (int k = lane; k < N_REC; k += 32)
    s += (float)zr[k] * wr[k];
  #pragma unroll
  for (int off = 16; off; off >>= 1)
    s += __shfl_xor(s, off, 32);
  if (lane == 0) {
    const float vn = KAPPAv * vo[b * N_OUT + o] + s;
    vo[b * N_OUT + o] = vn;
    outp[b * N_OUT + o] = vn;
  }
}

// ---- one-time conversion / init kernels ----
__global__ void cvt_x(const float* __restrict__ x, __bf16* __restrict__ xb, int nt) {
  const int total = nt * N_B * N_INP;
  for (int i = blockIdx.x * blockDim.x + threadIdx.x; i < total;
       i += gridDim.x * blockDim.x) {
    const int k  = i % N_INP;
    const int bt = i / N_INP;                 // t*128 + b
    xb[i] = (__bf16)((k < N_IN) ? x[bt * N_IN + k] : 0.f);
  }
}
__global__ void cvt_win(const float* __restrict__ w, __bf16* __restrict__ wb) {
  const int total = N_REC * N_INP;
  for (int i = blockIdx.x * blockDim.x + threadIdx.x; i < total;
       i += gridDim.x * blockDim.x) {
    const int k = i % N_INP;
    const int j = i / N_INP;
    wb[i] = (__bf16)((k < N_IN) ? w[j * N_IN + k] : 0.f);
  }
}
__global__ void cvt_wrec(const float* __restrict__ w, __bf16* __restrict__ wb) {
  const int total = N_REC * N_REC;
  for (int i = blockIdx.x * blockDim.x + threadIdx.x; i < total;
       i += gridDim.x * blockDim.x) {
    const int k = i % N_REC;
    const int j = i / N_REC;
    wb[i] = (__bf16)((j == k) ? 0.f : w[i]);
  }
}

// in-place softmax over last dim (20); one row per thread
__global__ void softmax_rows(float* __restrict__ p, int nrows) {
  const int r = blockIdx.x * blockDim.x + threadIdx.x;
  if (r >= nrows) return;
  float* row = p + (size_t)r * N_OUT;
  float mx = row[0];
  #pragma unroll
  for (int i = 1; i < N_OUT; ++i) mx = fmaxf(mx, row[i]);
  float e[N_OUT]; float s = 0.f;
  #pragma unroll
  for (int i = 0; i < N_OUT; ++i) { e[i] = __expf(row[i] - mx); s += e[i]; }
  const float inv = 1.0f / s;
  #pragma unroll
  for (int i = 0; i < N_OUT; ++i) row[i] = e[i] * inv;
}

extern "C" void kernel_launch(void* const* d_in, const int* in_sizes, int n_in,
                              void* d_out, int out_size, void* d_ws, size_t ws_size,
                              hipStream_t stream) {
  (void)in_sizes; (void)n_in; (void)out_size; (void)ws_size;
  const float* x     = (const float*)d_in[0];
  /* d_in[1] = yt (unused when do_training==0) */
  const float* w_in  = (const float*)d_in[2];
  const float* w_rec = (const float*)d_in[3];
  const float* w_out = (const float*)d_in[4];
  /* d_in[5] = do_training (0) */

  const int NTm1 = N_T - 1;   // 499 active steps
  char* ws = (char*)d_ws;
  size_t off = 0;
  auto carve = [&](size_t bytes) -> char* {
    char* p = ws + off;
    off = (off + bytes + 255) & ~(size_t)255;
    return p;
  };
  __bf16* xb    = (__bf16*)carve((size_t)NTm1 * N_B * N_INP * 2);  // ~90 MB
  __bf16* winb  = (__bf16*)carve((size_t)N_REC * N_INP * 2);
  __bf16* wrecb = (__bf16*)carve((size_t)N_REC * N_REC * 2);
  float*  v     = (float*) carve((size_t)N_B * N_REC * 4);
  __bf16* z0    = (__bf16*)carve((size_t)N_B * N_REC * 2);
  __bf16* z1    = (__bf16*)carve((size_t)N_B * N_REC * 2);
  float*  vo    = (float*) carve((size_t)N_B * N_OUT * 4);

  // zero state + t=0 output slice (graph-capture-safe stream memsets)
  hipMemsetAsync(v,  0, (size_t)N_B * N_REC * 4, stream);
  hipMemsetAsync(z0, 0, (size_t)N_B * N_REC * 2, stream);
  hipMemsetAsync(z1, 0, (size_t)N_B * N_REC * 2, stream);
  hipMemsetAsync(vo, 0, (size_t)N_B * N_OUT * 4, stream);
  hipMemsetAsync(d_out, 0, (size_t)N_B * N_OUT * 4, stream);

  cvt_x   <<<2048, 256, 0, stream>>>(x, xb, NTm1);
  cvt_win <<<512,  256, 0, stream>>>(w_in, winb);
  cvt_wrec<<<1024, 256, 0, stream>>>(w_rec, wrecb);

  __bf16* zc = z0;
  __bf16* zn = z1;
  float* out = (float*)d_out;
  for (int t = 0; t < NTm1; ++t) {
    srnn_step<<<N_REC / 16, 256, 0, stream>>>(
        xb + (size_t)t * N_B * N_INP, winb, wrecb, zc, v, zn);
    srnn_out<<<N_B, 640, 0, stream>>>(
        zn, w_out, vo, out + (size_t)(t + 1) * N_B * N_OUT);
    __bf16* tmp = zc; zc = zn; zn = tmp;
  }
  softmax_rows<<<(N_T * N_B + 255) / 256, 256, 0, stream>>>(out, N_T * N_B);
}
